// PerformerWithAttentionPool_36842229465473
// MI455X (gfx1250) — matrix operand order repaired and checked by reading, not compiled, not run on hardware
//
#include <hip/hip_runtime.h>
#include <hip/hip_bf16.h>
#include <math.h>

typedef _Float16 half_t;
typedef half_t v16h __attribute__((ext_vector_type(16)));
typedef half_t v8h  __attribute__((ext_vector_type(8)));
typedef float  v8f  __attribute__((ext_vector_type(8)));

#define cB     2
#define cN     16384
#define cIN    150
#define cMID   150
#define cL     4
#define cH     6
#define cDH    128
#define cINNER 768
#define cM     256
#define cFF    600
#define cED    1024
#define cOT    77
#define cHP    8
#define cHD    128
#define cPD    9600

struct GemmP {
  const float* __restrict__ A; long long Asm, Ask, Abo, Abi;
  const float* __restrict__ B; long long Bsk, Bsn, Bbo, Bbi;
  float*       __restrict__ C; long long Csm, Cbo, Cbi;
  int M, N, K, binner;
  const float* __restrict__ bias;
  const float* __restrict__ rowscale; long long rs_bo, rs_bi;
  float alpha;
  int flags; // 1 = accumulate into C, 2 = exact GELU
};

#define BM 128
#define BN 128
#define BK 32
#define LDA_S 40
#define LDB_S 40

__device__ __forceinline__ int imin(int a, int b) { return a < b ? a : b; }

__global__ __launch_bounds__(256) void gemm_wmma_kernel(GemmP p) {
  __shared__ __align__(32) half_t As[BM * LDA_S];
  __shared__ __align__(32) half_t Bs[BN * LDB_S];

  const int tid  = threadIdx.x;
  const int lane = tid & 31;

  const int z  = blockIdx.z;
  const int zo = z / p.binner;
  const int zi = z % p.binner;
  const float* __restrict__ Ap = p.A + (long long)zo * p.Abo + (long long)zi * p.Abi;
  const float* __restrict__ Bp = p.B + (long long)zo * p.Bbo + (long long)zi * p.Bbi;
  float*       __restrict__ Cp = p.C + (long long)zo * p.Cbo + (long long)zi * p.Cbi;
  const float* rs = p.rowscale ? (p.rowscale + (long long)zo * p.rs_bo + (long long)zi * p.rs_bi)
                               : (const float*)nullptr;

  const int tm = blockIdx.y * BM;
  const int tn = blockIdx.x * BN;
  const int wid = tid >> 5;
  const int wm = (wid >> 1) * 32;
  const int wn = (wid & 1) * 64;

  v8f acc[2][4];
#pragma unroll
  for (int i = 0; i < 2; ++i)
#pragma unroll
    for (int j = 0; j < 4; ++j)
#pragma unroll
      for (int e = 0; e < 8; ++e) acc[i][j][e] = 0.0f;

  const bool a_kfast = (p.Ask == 1);
  const bool b_kfast = (p.Bsk == 1);
  const int  hi  = lane >> 4;
  const int  l15 = lane & 15;

  // pair-path coordinates (unit-stride K): thread -> row r0+16i, kk..kk+1
  const int pr0 = tid >> 4;
  const int pkk = (tid & 15) * 2;
  const bool tileA_full = (tm + BM <= p.M);
  const bool tileB_full = (tn + BN <= p.N);

  // ---- loaders: global -> registers (branch-free on lanes; uniform fast path) ----
  auto loadA = [&](int k0, float* v) {
    const bool kfull = (k0 + BK <= p.K);
    if (a_kfast) {
      if (tileA_full && kfull) {
#pragma unroll
        for (int i = 0; i < 8; ++i) {
          const float* q = Ap + (long long)(tm + pr0 + 16 * i) * p.Asm + (k0 + pkk);
          v[2 * i] = q[0]; v[2 * i + 1] = q[1];
        }
      } else {
#pragma unroll
        for (int i = 0; i < 8; ++i) {
          const int gm = tm + pr0 + 16 * i;
          const int gk = k0 + pkk;
          const long long base = (long long)imin(gm, p.M - 1) * p.Asm;
          const float x0 = Ap[base + imin(gk, p.K - 1)];
          const float x1 = Ap[base + imin(gk + 1, p.K - 1)];
          v[2 * i]     = (gm < p.M && gk     < p.K) ? x0 : 0.0f;
          v[2 * i + 1] = (gm < p.M && gk + 1 < p.K) ? x1 : 0.0f;
        }
      }
    } else {
      // row-fast mapping (coalesced when Asm==1): e = tid + i*256; row=e&127; kk=e>>7
      if (tileA_full && kfull) {
#pragma unroll
        for (int i = 0; i < 16; ++i) {
          const int e = tid + i * 256;
          v[i] = Ap[(long long)(tm + (e & 127)) * p.Asm + (long long)(k0 + (e >> 7)) * p.Ask];
        }
      } else {
#pragma unroll
        for (int i = 0; i < 16; ++i) {
          const int e = tid + i * 256;
          const int gm = tm + (e & 127), gk = k0 + (e >> 7);
          const float x = Ap[(long long)imin(gm, p.M - 1) * p.Asm + (long long)imin(gk, p.K - 1) * p.Ask];
          v[i] = (gm < p.M && gk < p.K) ? x : 0.0f;
        }
      }
    }
  };
  auto storeA = [&](const float* v) {
    if (a_kfast) {
#pragma unroll
      for (int i = 0; i < 8; ++i) {
        union { half_t h[2]; unsigned u; } pk;
        pk.h[0] = (half_t)v[2 * i]; pk.h[1] = (half_t)v[2 * i + 1];
        *(unsigned*)&As[(pr0 + 16 * i) * LDA_S + pkk] = pk.u;
      }
    } else {
#pragma unroll
      for (int i = 0; i < 16; ++i) {
        const int e = tid + i * 256;
        As[(e & 127) * LDA_S + (e >> 7)] = (half_t)v[i];
      }
    }
  };
  auto loadB = [&](int k0, float* v) {
    const bool kfull = (k0 + BK <= p.K);
    if (b_kfast) {
      if (tileB_full && kfull) {
#pragma unroll
        for (int i = 0; i < 8; ++i) {
          const float* q = Bp + (long long)(tn + pr0 + 16 * i) * p.Bsn + (k0 + pkk);
          v[2 * i] = q[0]; v[2 * i + 1] = q[1];
        }
      } else {
#pragma unroll
        for (int i = 0; i < 8; ++i) {
          const int gn = tn + pr0 + 16 * i;
          const int gk = k0 + pkk;
          const long long base = (long long)imin(gn, p.N - 1) * p.Bsn;
          const float x0 = Bp[base + imin(gk, p.K - 1)];
          const float x1 = Bp[base + imin(gk + 1, p.K - 1)];
          v[2 * i]     = (gn < p.N && gk     < p.K) ? x0 : 0.0f;
          v[2 * i + 1] = (gn < p.N && gk + 1 < p.K) ? x1 : 0.0f;
        }
      }
    } else {
      // col-fast mapping (coalesced when Bsn==1): e = tid + i*256; col=e&127; kk=e>>7
      if (tileB_full && kfull) {
#pragma unroll
        for (int i = 0; i < 16; ++i) {
          const int e = tid + i * 256;
          v[i] = Bp[(long long)(k0 + (e >> 7)) * p.Bsk + (long long)(tn + (e & 127)) * p.Bsn];
        }
      } else {
#pragma unroll
        for (int i = 0; i < 16; ++i) {
          const int e = tid + i * 256;
          const int gn = tn + (e & 127), gk = k0 + (e >> 7);
          const float x = Bp[(long long)imin(gk, p.K - 1) * p.Bsk + (long long)imin(gn, p.N - 1) * p.Bsn];
          v[i] = (gn < p.N && gk < p.K) ? x : 0.0f;
        }
      }
    }
  };
  auto storeB = [&](const float* v) {
    if (b_kfast) {
#pragma unroll
      for (int i = 0; i < 8; ++i) {
        union { half_t h[2]; unsigned u; } pk;
        pk.h[0] = (half_t)v[2 * i]; pk.h[1] = (half_t)v[2 * i + 1];
        *(unsigned*)&Bs[(pr0 + 16 * i) * LDB_S + pkk] = pk.u;
      }
    } else {
#pragma unroll
      for (int i = 0; i < 16; ++i) {
        const int e = tid + i * 256;
        Bs[(e & 127) * LDB_S + (e >> 7)] = (half_t)v[i];
      }
    }
  };

  // ---- pipelined main loop: next-tile global loads overlap WMMA ----
  float va[16], vb[16];
  loadA(0, va);
  loadB(0, vb);

  for (int k0 = 0; k0 < p.K; k0 += BK) {
    storeA(va);
    storeB(vb);
    __syncthreads();

    const bool more = (k0 + BK) < p.K;
    float na[16], nb[16];
    if (more) { loadA(k0 + BK, na); loadB(k0 + BK, nb); }

    v16h bf[4];
#pragma unroll
    for (int sn = 0; sn < 4; ++sn) {
      const half_t* bp = &Bs[(wn + sn * 16 + l15) * LDB_S + hi * 16];
      v8h b0 = *(const v8h*)bp;
      v8h b1 = *(const v8h*)(bp + 8);
      bf[sn] = __builtin_shufflevector(b0, b1, 0,1,2,3,4,5,6,7,8,9,10,11,12,13,14,15);
    }
#pragma unroll
    for (int sm = 0; sm < 2; ++sm) {
      const half_t* ap = &As[(wm + sm * 16 + l15) * LDA_S + hi * 16];
      v8h a0 = *(const v8h*)ap;
      v8h a1 = *(const v8h*)(ap + 8);
      v16h af = __builtin_shufflevector(a0, a1, 0,1,2,3,4,5,6,7,8,9,10,11,12,13,14,15);
#pragma unroll
      for (int sn = 0; sn < 4; ++sn) {
        acc[sm][sn] = __builtin_amdgcn_wmma_f32_16x16x32_f16(
            false, af, false, bf[sn], (short)0, acc[sm][sn], false, false);
      }
    }
    __syncthreads();

    if (more) {
#pragma unroll
      for (int i = 0; i < 16; ++i) { va[i] = na[i]; vb[i] = nb[i]; }
    }
  }

  // ---- epilogue ----
#pragma unroll
  for (int sm = 0; sm < 2; ++sm)
#pragma unroll
    for (int sn = 0; sn < 4; ++sn)
#pragma unroll
      for (int j = 0; j < 8; ++j) {
        const int row = tm + wm + sm * 16 + hi * 8 + j;
        const int col = tn + wn + sn * 16 + l15;
        if (row < p.M && col < p.N) {
          float v = acc[sm][sn][j] * p.alpha;
          if (p.bias) v += p.bias[col];
          if (p.flags & 2) v = 0.5f * v * (1.0f + erff(v * 0.70710678118654752f));
          if (rs) v *= rs[row];
          const long long ci = (long long)row * p.Csm + col;
          if (p.flags & 1) Cp[ci] += v; else Cp[ci] = v;
        }
      }
}

__device__ __forceinline__ float breduce_sum(float v, float* sm) {
  const int t = threadIdx.x;
  sm[t] = v; __syncthreads();
#pragma unroll
  for (int s = 128; s > 0; s >>= 1) { if (t < s) sm[t] += sm[t + s]; __syncthreads(); }
  const float r = sm[0]; __syncthreads(); return r;
}
__device__ __forceinline__ float breduce_max(float v, float* sm) {
  const int t = threadIdx.x;
  sm[t] = v; __syncthreads();
#pragma unroll
  for (int s = 128; s > 0; s >>= 1) { if (t < s) sm[t] = fmaxf(sm[t], sm[t + s]); __syncthreads(); }
  const float r = sm[0]; __syncthreads(); return r;
}

__global__ __launch_bounds__(256) void layernorm_kernel(const float* __restrict__ x,
                                                        const float* __restrict__ g,
                                                        const float* __restrict__ b,
                                                        float* __restrict__ out,
                                                        int D, float eps) {
  __shared__ float sm[256];
  const long long row = blockIdx.x;
  const float* xr = x + row * (long long)D;
  float s = 0.0f, s2 = 0.0f;
  for (int i = threadIdx.x; i < D; i += 256) { const float v = xr[i]; s += v; s2 += v * v; }
  s  = breduce_sum(s,  sm);
  s2 = breduce_sum(s2, sm);
  const float m   = s / (float)D;
  const float var = s2 / (float)D - m * m;
  const float inv = rsqrtf(var + eps);
  float* orow = out + row * (long long)D;
  for (int i = threadIdx.x; i < D; i += 256) orow[i] = (xr[i] - m) * inv * g[i] + b[i];
}

__global__ __launch_bounds__(256) void addpos_kernel(float* __restrict__ h,
                                                     const float* __restrict__ pos) {
  const unsigned i = blockIdx.x * 256u + threadIdx.x;
  if (i < 4915200u) {
    const unsigned nc = (i >= 2457600u) ? (i - 2457600u) : i;
    h[i] += pos[nc];
  }
}

__global__ __launch_bounds__(256) void rowmax_kernel(const float* __restrict__ P,
                                                     float* __restrict__ rmax) {
  __shared__ float sm[256];
  const long long r = blockIdx.x;
  const float m = breduce_max(P[r * cM + threadIdx.x], sm);
  if (threadIdx.x == 0) rmax[r] = m;
}

__global__ __launch_bounds__(256) void colmax_kernel(const float* __restrict__ rmax,
                                                     float* __restrict__ stab, int count) {
  __shared__ float sm[256];
  const int z = blockIdx.x;
  float m = -3.4e38f;
  for (int i = threadIdx.x; i < count; i += 256) m = fmaxf(m, rmax[(long long)z * count + i]);
  m = breduce_max(m, sm);
  if (threadIdx.x == 0) stab[z] = m;
}

__global__ __launch_bounds__(256) void featurize_kernel(float* __restrict__ P,
                                                        const float* __restrict__ qk,
                                                        const float* __restrict__ stab,
                                                        const float* __restrict__ ksum,
                                                        float* __restrict__ dinv,
                                                        int is_query, float dn, float ratio) {
  __shared__ float sm[256];
  const unsigned r = blockIdx.x;
  const int tid = threadIdx.x;
  const unsigned zz = r >> 14;          // r / cN
  const int n  = (int)(r & 16383u);     // r % cN
  const int b  = (int)(zz / cH);
  const int hh = (int)(zz % cH);
  const float* qrow = qk + ((long long)b * cN + n) * cINNER + (long long)hh * cDH;
  const float qv = (tid < cDH) ? qrow[tid] : 0.0f;
  const float ss = breduce_sum(qv * qv, sm);
  const float diag = ss * 0.5f * dn * dn;
  const float ddv = P[(long long)r * cM + tid];
  float st;
  if (is_query) st = breduce_max(ddv, sm);
  else          st = stab[zz];
  const float qp = ratio * (expf(ddv - diag - st) + 1e-4f);
  P[(long long)r * cM + tid] = qp;
  if (is_query) {
    const float d = breduce_sum(qp * ksum[(long long)zz * cM + tid], sm);
    if (tid == 0) dinv[r] = 1.0f / d;
  }
}

__global__ __launch_bounds__(256) void zero_kernel(float* p, long long n) {
  const long long i = (long long)blockIdx.x * 256 + threadIdx.x;
  if (i < n) p[i] = 0.0f;
}

__global__ __launch_bounds__(256) void ksum_kernel(const float* __restrict__ P,
                                                   float* __restrict__ ksum) {
  const int z = blockIdx.x, chunk = blockIdx.y, m = threadIdx.x;
  const long long base = (long long)z * cN * cM + (long long)chunk * 256 * cM;
  float s = 0.0f;
  for (int n = 0; n < 256; ++n) s += P[base + (long long)n * cM + m];
  atomicAdd(&ksum[(long long)z * cM + m], s);
}

__global__ __launch_bounds__(256) void patches_kernel(const float* __restrict__ h,
                                                      float* __restrict__ patches) {
  const unsigned i = blockIdx.x * 256u + threadIdx.x;
  if (i >= 4915200u) return;
  const unsigned k = i % cPD;
  const unsigned t = i / cPD;
  const unsigned pidx = t & 255u;
  const unsigned b = t >> 8;
  const unsigned c = k >> 6, rem = k & 63u, py = rem >> 3, px = rem & 7u;
  const unsigned gy = pidx >> 4, gx = pidx & 15u;
  const unsigned n = (gy * 8 + py) * 128 + gx * 8 + px;
  patches[i] = h[((long long)b * cN + n) * cMID + c];
}

__global__ __launch_bounds__(256) void softmax256_kernel(float* __restrict__ s) {
  __shared__ float sm[256];
  const long long r = blockIdx.x;
  const float v = s[r * 256 + threadIdx.x];
  const float mx = breduce_max(v, sm);
  const float e = expf(v - mx);
  const float su = breduce_sum(e, sm);
  s[r * 256 + threadIdx.x] = e / su;
}

static void gemm(hipStream_t st,
                 const float* A, long long Asm, long long Ask, long long Abo, long long Abi,
                 const float* B, long long Bsk, long long Bsn, long long Bbo, long long Bbi,
                 float* C, long long Csm, long long Cbo, long long Cbi,
                 int M, int N, int K, int nbatch, int binner,
                 const float* bias, const float* rowscale, long long rs_bo, long long rs_bi,
                 float alpha, int flags) {
  GemmP p;
  p.A = A; p.Asm = Asm; p.Ask = Ask; p.Abo = Abo; p.Abi = Abi;
  p.B = B; p.Bsk = Bsk; p.Bsn = Bsn; p.Bbo = Bbo; p.Bbi = Bbi;
  p.C = C; p.Csm = Csm; p.Cbo = Cbo; p.Cbi = Cbi;
  p.M = M; p.N = N; p.K = K; p.binner = binner;
  p.bias = bias; p.rowscale = rowscale; p.rs_bo = rs_bo; p.rs_bi = rs_bi;
  p.alpha = alpha; p.flags = flags;
  dim3 g((N + BN - 1) / BN, (M + BM - 1) / BM, nbatch);
  gemm_wmma_kernel<<<g, dim3(256), 0, st>>>(p);
}

extern "C" void kernel_launch(void* const* d_in, const int* in_sizes, int n_in,
                              void* d_out, int out_size, void* d_ws, size_t ws_size,
                              hipStream_t stream) {
  (void)in_sizes; (void)out_size;
  if (n_in < 30) return;

  const float* x        = (const float*)d_in[0];
  const float* proj1_w  = (const float*)d_in[1];
  const float* proj1_b  = (const float*)d_in[2];
  const float* pos      = (const float*)d_in[3];
  const float* ln1_g    = (const float*)d_in[4];
  const float* ln1_b    = (const float*)d_in[5];
  const float* wq       = (const float*)d_in[6];
  const float* bq       = (const float*)d_in[7];
  const float* wk       = (const float*)d_in[8];
  const float* bk       = (const float*)d_in[9];
  const float* wv       = (const float*)d_in[10];
  const float* bv       = (const float*)d_in[11];
  const float* wo       = (const float*)d_in[12];
  const float* bo       = (const float*)d_in[13];
  const float* projm    = (const float*)d_in[14];
  const float* ln2_g    = (const float*)d_in[15];
  const float* ln2_b    = (const float*)d_in[16];
  const float* ffw1     = (const float*)d_in[17];
  const float* ffb1     = (const float*)d_in[18];
  const float* ffw2     = (const float*)d_in[19];
  const float* ffb2     = (const float*)d_in[20];
  const float* le_w     = (const float*)d_in[21];
  const float* le_b     = (const float*)d_in[22];
  const float* qt       = (const float*)d_in[23];
  const float* mha_in_w = (const float*)d_in[24];
  const float* mha_in_b = (const float*)d_in[25];
  const float* mha_out_w= (const float*)d_in[26];
  const float* mha_out_b= (const float*)d_in[27];
  const float* lnf_g    = (const float*)d_in[28];
  const float* lnf_b    = (const float*)d_in[29];

  const long long O_H    = 0;
  const long long O_Y    = O_H    + 4915200LL;
  const long long O_Q    = O_Y    + 4915200LL;
  const long long O_K    = O_Q    + 25165824LL;
  const long long O_V    = O_K    + 25165824LL;
  const long long O_P    = O_V    + 25165824LL;
  const long long O_CTX  = O_P    + 50331648LL;
  const long long O_KSUM = O_CTX  + 393216LL;
  const long long O_STAB = O_KSUM + 3072LL;
  const long long O_RMAX = O_STAB + 16LL;
  const long long O_DINV = O_RMAX + 196608LL;
  const long long O_KV   = O_DINV + 196608LL;
  const long long O_QH   = O_KV   + 524288LL;
  const long long O_KH   = O_QH   + 78848LL;
  const long long O_VH   = O_KH   + 524288LL;
  const long long O_SC   = O_VH   + 524288LL;
  const long long TOTALF = O_SC   + 315392LL;
  if (ws_size < (size_t)TOTALF * sizeof(float)) return;

  float* ws   = (float*)d_ws;
  float* h    = ws + O_H;
  float* y    = ws + O_Y;
  float* qb   = ws + O_Q;
  float* kb   = ws + O_K;
  float* vb   = ws + O_V;
  float* P    = ws + O_P;
  float* ctx  = ws + O_CTX;
  float* ksum = ws + O_KSUM;
  float* stab = ws + O_STAB;
  float* rmax = ws + O_RMAX;
  float* dinv = ws + O_DINV;
  float* kv   = ws + O_KV;
  float* qh   = ws + O_QH;
  float* kh   = ws + O_KH;
  float* vh   = ws + O_VH;
  float* sc   = ws + O_SC;
  float* po   = ctx;   // reuse (ctx free after layers)
  float* preln= kb;    // reuse
  float* pat  = y;     // reuse

  const int BT = cB * cN;
  const float dn    = 0.29730177875068026f;
  const float ratio = 0.0625f;
  const long long qkvBO = (long long)cN * cINNER;
  const long long pBO   = (long long)cH * cN * cM;
  const long long pBI   = (long long)cN * cM;

  gemm(stream, x, cIN, 1, 0, 0, proj1_w, cMID, 1, 0, 0, h, cMID, 0, 0,
       BT, cMID, cIN, 1, 1, proj1_b, nullptr, 0, 0, 1.0f, 0);
  addpos_kernel<<<(4915200 + 255) / 256, 256, 0, stream>>>(h, pos);

  for (int l = 0; l < cL; ++l) {
    layernorm_kernel<<<BT, 256, 0, stream>>>(h, ln1_g + l * cMID, ln1_b + l * cMID, y, cMID, 1e-5f);
    gemm(stream, y, cMID, 1, 0, 0, wq + (long long)l * cMID * cINNER, cINNER, 1, 0, 0,
         qb, cINNER, 0, 0, BT, cINNER, cMID, 1, 1, bq + (long long)l * cINNER, nullptr, 0, 0, 1.0f, 0);
    gemm(stream, y, cMID, 1, 0, 0, wk + (long long)l * cMID * cINNER, cINNER, 1, 0, 0,
         kb, cINNER, 0, 0, BT, cINNER, cMID, 1, 1, bk + (long long)l * cINNER, nullptr, 0, 0, 1.0f, 0);
    gemm(stream, y, cMID, 1, 0, 0, wv + (long long)l * cMID * cINNER, cINNER, 1, 0, 0,
         vb, cINNER, 0, 0, BT, cINNER, cMID, 1, 1, bv + (long long)l * cINNER, nullptr, 0, 0, 1.0f, 0);

    const float* pm = projm + (long long)l * cM * cDH;

    gemm(stream, kb, cINNER, 1, qkvBO, cDH, pm, 1, cDH, 0, 0,
         P, cM, pBO, pBI, cN, cM, cDH, cB * cH, cH, nullptr, nullptr, 0, 0, dn, 0);
    rowmax_kernel<<<cB * cH * cN, 256, 0, stream>>>(P, rmax);
    colmax_kernel<<<cB * cH, 256, 0, stream>>>(rmax, stab, cN);
    featurize_kernel<<<cB * cH * cN, 256, 0, stream>>>(P, kb, stab, nullptr, nullptr, 0, dn, ratio);
    zero_kernel<<<(cB * cH * cM + 255) / 256, 256, 0, stream>>>(ksum, (long long)cB * cH * cM);
    ksum_kernel<<<dim3(cB * cH, cN / 256), 256, 0, stream>>>(P, ksum);

    gemm(stream, P, 1, cM, pBO, pBI, vb, cINNER, 1, qkvBO, cDH,
         ctx, cDH, (long long)cH * cM * cDH, (long long)cM * cDH,
         cM, cDH, cN, cB * cH, cH, nullptr, nullptr, 0, 0, 1.0f, 0);

    gemm(stream, qb, cINNER, 1, qkvBO, cDH, pm, 1, cDH, 0, 0,
         P, cM, pBO, pBI, cN, cM, cDH, cB * cH, cH, nullptr, nullptr, 0, 0, dn, 0);
    featurize_kernel<<<cB * cH * cN, 256, 0, stream>>>(P, qb, nullptr, ksum, dinv, 1, dn, ratio);

    gemm(stream, P, cM, 1, pBO, pBI, ctx, cDH, 1, (long long)cH * cM * cDH, (long long)cM * cDH,
         kb, cINNER, qkvBO, cDH, cN, cDH, cM, cB * cH, cH,
         nullptr, dinv, (long long)cH * cN, (long long)cN, 1.0f, 0);

    gemm(stream, kb, cINNER, 1, 0, 0, wo + (long long)l * cINNER * cMID, cMID, 1, 0, 0,
         h, cMID, 0, 0, BT, cMID, cINNER, 1, 1, bo + (long long)l * cMID, nullptr, 0, 0, 1.0f, 1);

    layernorm_kernel<<<BT, 256, 0, stream>>>(h, ln2_g + l * cMID, ln2_b + l * cMID, y, cMID, 1e-5f);
    gemm(stream, y, cMID, 1, 0, 0, ffw1 + (long long)l * cMID * cFF, cFF, 1, 0, 0,
         qb, cFF, 0, 0, BT, cFF, cMID, 1, 1, ffb1 + (long long)l * cFF, nullptr, 0, 0, 1.0f, 2);
    gemm(stream, qb, cFF, 1, 0, 0, ffw2 + (long long)l * cFF * cMID, cMID, 1, 0, 0,
         h, cMID, 0, 0, BT, cMID, cFF, 1, 1, ffb2 + (long long)l * cMID, nullptr, 0, 0, 1.0f, 1);
  }

  patches_kernel<<<(4915200 + 255) / 256, 256, 0, stream>>>(h, pat);
  gemm(stream, pat, cPD, 1, 0, 0, le_w, cED, 1, 0, 0, kv, cED, 0, 0,
       cB * 256, cED, cPD, 1, 1, le_b, nullptr, 0, 0, 1.0f, 0);

  gemm(stream, qt, cED, 1, 0, 0, mha_in_w, 1, cED, 0, 0, qh, cED, 0, 0,
       cOT, cED, cED, 1, 1, mha_in_b, nullptr, 0, 0, 1.0f, 0);
  gemm(stream, kv, cED, 1, 0, 0, mha_in_w + (long long)cED * cED, 1, cED, 0, 0, kh, cED, 0, 0,
       cB * 256, cED, cED, 1, 1, mha_in_b + cED, nullptr, 0, 0, 1.0f, 0);
  gemm(stream, kv, cED, 1, 0, 0, mha_in_w + 2LL * cED * cED, 1, cED, 0, 0, vh, cED, 0, 0,
       cB * 256, cED, cED, 1, 1, mha_in_b + 2 * cED, nullptr, 0, 0, 1.0f, 0);

  gemm(stream, qh, cED, 1, 0, cHD, kh, 1, cED, 256LL * cED, cHD,
       sc, 256, (long long)cHP * cOT * 256, (long long)cOT * 256,
       cOT, 256, cHD, cB * cHP, cHP, nullptr, nullptr, 0, 0, 0.088388347648318447f, 0);
  softmax256_kernel<<<cB * cHP * cOT, 256, 0, stream>>>(sc);

  gemm(stream, sc, 256, 1, (long long)cHP * cOT * 256, (long long)cOT * 256,
       vh, cED, 1, 256LL * cED, cHD,
       po, cED, (long long)cOT * cED, cHD,
       cOT, cHD, 256, cB * cHP, cHP, nullptr, nullptr, 0, 0, 1.0f, 0);

  gemm(stream, po, cED, 1, 0, 0, mha_out_w, 1, cED, 0, 0, preln, cED, 0, 0,
       cB * cOT, cED, cED, 1, 1, mha_out_b, nullptr, 0, 0, 1.0f, 0);
  layernorm_kernel<<<cB * cOT, 256, 0, stream>>>(preln, lnf_g, lnf_b, (float*)d_out, cED, 1e-5f);
}
